// MultiHeadAttention_73014444032543
// MI455X (gfx1250) — compile-verified, matrix operand order checked by reading, not run on hardware
//
#include <hip/hip_runtime.h>
#include <hip/hip_bf16.h>

// ---------------------------------------------------------------------------
// MI455X (gfx1250) multi-head causal attention, bf16 WMMA pipeline.
// B=2, T=2048, C=1024, H=16, HS=64.
// Round 2: register blocking in both GEMMs to raise WMMA : VMEM issue ratio.
// ---------------------------------------------------------------------------

typedef __bf16 bf16_t;
typedef __attribute__((ext_vector_type(16))) __bf16 v16bf;
typedef __attribute__((ext_vector_type(8)))  __bf16 v8bf;
typedef __attribute__((ext_vector_type(8)))  float  v8f;

constexpr int NB = 2;
constexpr int NT = 2048;
constexpr int NC = 1024;
constexpr int NH = 16;
constexpr int ND = 64;   // head size

// Load one 16-bit A/B WMMA fragment (16 bf16 per lane) for a 32-deep K chunk.
// Per ISA layout: element e of the v16bf holds K = 16*(e>=8) + koff + (e&7),
// where koff = 8*(lane>=16). rowbase points at (row, kchunk_base) and rows are
// contiguous in K, so this is two 16-byte loads.
__device__ __forceinline__ v16bf load_frag(const bf16_t* rowbase, int koff) {
  v8bf lo = *(const v8bf*)(rowbase + koff);
  v8bf hi = *(const v8bf*)(rowbase + 16 + koff);
  v16bf f;
#pragma unroll
  for (int i = 0; i < 8; ++i) { f[i] = lo[i]; f[i + 8] = hi[i]; }
  return f;
}

__device__ __forceinline__ v8f wmma_bf16(v16bf a, v16bf b, v8f c) {
  // (neg_a, A, neg_b, B, c_mod, C, reuse_a, reuse_b)
  return __builtin_amdgcn_wmma_f32_16x16x32_bf16(false, a, false, b, (short)0,
                                                 c, false, false);
}

// ---------------------------------------------------------------------------
// Kernel 0a: elementwise fp32 -> bf16
// ---------------------------------------------------------------------------
__global__ void cvt_bf16(const float* __restrict__ in, bf16_t* __restrict__ out,
                         int n) {
  int i = blockIdx.x * blockDim.x + threadIdx.x;
  if (i < n) out[i] = (bf16_t)in[i];
}

// ---------------------------------------------------------------------------
// Kernel 0b: fp32 [nmat][rows][cols] -> bf16 [nmat][cols][rows]
// (wq/wk/wv: [H][C][HS] -> [H][HS][C], so B-fragments read contiguous K)
// ---------------------------------------------------------------------------
__global__ void cvt_bf16_tr(const float* __restrict__ in,
                            bf16_t* __restrict__ out, int nmat, int rows,
                            int cols) {
  int i = blockIdx.x * blockDim.x + threadIdx.x;
  int per = rows * cols;
  int total = nmat * per;
  if (i < total) {
    int m = i / per;
    int rem = i - m * per;
    int r = rem / cols;
    int c = rem - r * cols;
    out[(size_t)m * per + (size_t)c * rows + r] = (bf16_t)in[i];
  }
}

// ---------------------------------------------------------------------------
// Kernel 1: QKV projection. One wave per (b,h, 32-row t-tile, d-tile).
// 2 M-tiles x {q,k,v}: the weight B-fragments are reused across both M-tiles,
// so each K-step is 5 fragment loads : 6 WMMAs.
//   q,k stored [B*H][T][HS] row-major bf16; v stored transposed [B*H][HS][T].
// Grid: 32 * 64 * 4 = 8192 waves = 1024 blocks of 256 threads.
// ---------------------------------------------------------------------------
__global__ void qkv_gemm(const bf16_t* __restrict__ xb,
                         const bf16_t* __restrict__ wqt,
                         const bf16_t* __restrict__ wkt,
                         const bf16_t* __restrict__ wvt,
                         bf16_t* __restrict__ qb, bf16_t* __restrict__ kb,
                         bf16_t* __restrict__ vtb) {
  const int lane = threadIdx.x & 31;
  const int wave = blockIdx.x * 8 + (threadIdx.x >> 5);
  const int id = wave & 3;          // d tile (HS/16 = 4)
  const int it = (wave >> 2) & 63;  // 32-row t tile (T/32 = 64)
  const int bh = wave >> 8;         // b*H + h (32)
  const int b = bh >> 4;
  const int h = bh & 15;
  const int t0 = it * 32;
  const int d0 = id * 16;
  const int lrow = lane & 15;
  const int half = lane >> 4;
  const int koff = half * 8;

  const bf16_t* arow0 = xb + ((size_t)(b * NT + t0 + lrow)) * NC;
  const bf16_t* arow1 = arow0 + (size_t)16 * NC;
  const bf16_t* bq = wqt + ((size_t)(h * ND + d0 + lrow)) * NC;
  const bf16_t* bk = wkt + ((size_t)(h * ND + d0 + lrow)) * NC;
  const bf16_t* bv = wvt + ((size_t)(h * ND + d0 + lrow)) * NC;

  v8f accq0 = {}, acck0 = {}, accv0 = {};
  v8f accq1 = {}, acck1 = {}, accv1 = {};
  for (int kk = 0; kk < NC; kk += 32) {
    __builtin_prefetch(arow0 + kk + 128, 0, 0);  // global_prefetch_b8
    v16bf a0 = load_frag(arow0 + kk, koff);
    v16bf a1 = load_frag(arow1 + kk, koff);
    v16bf fq = load_frag(bq + kk, koff);
    v16bf fk = load_frag(bk + kk, koff);
    v16bf fv = load_frag(bv + kk, koff);
    accq0 = wmma_bf16(a0, fq, accq0);
    accq1 = wmma_bf16(a1, fq, accq1);
    acck0 = wmma_bf16(a0, fk, acck0);
    acck1 = wmma_bf16(a1, fk, acck1);
    accv0 = wmma_bf16(a0, fv, accv0);
    accv1 = wmma_bf16(a1, fv, accv1);
  }

#pragma unroll
  for (int r = 0; r < 8; ++r) {
    const int t = t0 + r + 8 * half;  // C/D layout: M = r + 8*(lane>=16)
    const int d = d0 + lrow;          // N = lane & 15
    const size_t qi0 = ((size_t)bh * NT + t) * ND + d;
    const size_t qi1 = qi0 + (size_t)16 * ND;
    qb[qi0] = (bf16_t)accq0[r];
    kb[qi0] = (bf16_t)acck0[r];
    qb[qi1] = (bf16_t)accq1[r];
    kb[qi1] = (bf16_t)acck1[r];
    bf16_t* vcol = vtb + ((size_t)bh * ND + d) * NT + t;
    vcol[0] = (bf16_t)accv0[r];   // V transposed
    vcol[16] = (bf16_t)accv1[r];
  }
}

// ---------------------------------------------------------------------------
// Kernel 2: flash attention (causal, online softmax). One wave per
// (b,h, 16-query tile), streaming 32 keys per step.
// Grid: 32 * 128 = 4096 waves = 512 blocks of 256 threads.
// ---------------------------------------------------------------------------
__global__ void flash_attn(const bf16_t* __restrict__ qb,
                           const bf16_t* __restrict__ kb,
                           const bf16_t* __restrict__ vtb,
                           bf16_t* __restrict__ ob) {
  __shared__ __align__(16) bf16_t lds[8][16 * 32];  // P staging, 1KB per wave
  const int lane = threadIdx.x & 31;
  const int wslot = threadIdx.x >> 5;
  const int wave = blockIdx.x * 8 + wslot;
  const int it = wave & 127;
  const int bh = wave >> 7;
  const int b = bh >> 4;
  const int h = bh & 15;
  const int t0 = it * 16;
  const int lrow = lane & 15;
  const int half = lane >> 4;
  const int koff = half * 8;
  bf16_t* sl = &lds[wslot][0];

  // Q tile as two A-fragments (d = 0..31 and 32..63).
  const bf16_t* qrow = qb + ((size_t)bh * NT + t0 + lrow) * ND;
  const v16bf q0 = load_frag(qrow, koff);
  const v16bf q1 = load_frag(qrow + 32, koff);

  float m[8], lsum[8];
  v8f o[4] = {v8f{}, v8f{}, v8f{}, v8f{}};
#pragma unroll
  for (int r = 0; r < 8; ++r) { m[r] = -1e30f; lsum[r] = 0.0f; }

  const int t_hi = t0 + 15;
  for (int s0 = 0; s0 <= t_hi; s0 += 32) {  // uniform per wave (causal bound)
    // ---- S = (Q K^T) * scale for 32 keys: two 16-key column tiles ----
    const bf16_t* kr0 = kb + ((size_t)bh * NT + s0 + lrow) * ND;
    const bf16_t* kr1 = kr0 + 16 * ND;
    v16bf k0a = load_frag(kr0, koff);
    v16bf k0b = load_frag(kr0 + 32, koff);
    v16bf k1a = load_frag(kr1, koff);
    v16bf k1b = load_frag(kr1 + 32, koff);
    v8f sA = {}, sB = {};
    sA = wmma_bf16(q0, k0a, sA);
    sA = wmma_bf16(q1, k0b, sA);
    sB = wmma_bf16(q0, k1a, sB);
    sB = wmma_bf16(q1, k1b, sB);

    // ---- scale + causal mask (element (M,N): t = t0+M, s = s0(+16)+N) ----
    float pa[8], pb[8];
#pragma unroll
    for (int r = 0; r < 8; ++r) {
      const int t = t0 + r + 8 * half;
      float a = sA[r] * 0.125f;  // 1/sqrt(64)
      float c = sB[r] * 0.125f;
      if (s0 + lrow > t) a = -1e30f;
      if (s0 + 16 + lrow > t) c = -1e30f;
      pa[r] = a;
      pb[r] = c;
    }

    // ---- online softmax: per-row reductions across the 16 lanes of a half --
#pragma unroll
    for (int r = 0; r < 8; ++r) {
      float rmax = fmaxf(pa[r], pb[r]);
#pragma unroll
      for (int msk = 1; msk < 16; msk <<= 1)
        rmax = fmaxf(rmax, __shfl_xor(rmax, msk, 16));
      const float mn = fmaxf(m[r], rmax);
      const float sc = __expf(m[r] - mn);
      const float ea = __expf(pa[r] - mn);
      const float eb = __expf(pb[r] - mn);
      pa[r] = ea;
      pb[r] = eb;
      float rs = ea + eb;
#pragma unroll
      for (int msk = 1; msk < 16; msk <<= 1) rs += __shfl_xor(rs, msk, 16);
      lsum[r] = lsum[r] * sc + rs;
      m[r] = mn;
      o[0][r] *= sc;
      o[1][r] *= sc;
      o[2][r] *= sc;
      o[3][r] *= sc;
    }

    // ---- re-layout P: C/D layout -> A fragment via per-wave LDS tile ----
#pragma unroll
    for (int r = 0; r < 8; ++r) {
      const int M = r + 8 * half;
      sl[M * 32 + lrow] = (bf16_t)pa[r];
      sl[M * 32 + 16 + lrow] = (bf16_t)pb[r];
    }
    const v16bf pf = load_frag(sl + lrow * 32, koff);  // in-wave DS ordering

    // ---- O += P * V : V^T layout gives contiguous-K B fragments ----
    const bf16_t* vbase = vtb + (size_t)bh * ND * NT + s0;
#pragma unroll
    for (int nt = 0; nt < 4; ++nt) {
      v16bf vf = load_frag(vbase + (size_t)(nt * 16 + lrow) * NT, koff);
      o[nt] = wmma_bf16(pf, vf, o[nt]);
    }
  }

  // ---- normalize and write attention output, heads concatenated -> [B,T,C]
#pragma unroll
  for (int nt = 0; nt < 4; ++nt) {
#pragma unroll
    for (int r = 0; r < 8; ++r) {
      const int t = t0 + r + 8 * half;
      const int d = nt * 16 + lrow;
      ob[((size_t)(b * NT + t)) * NC + h * ND + d] =
          (bf16_t)(o[nt][r] / lsum[r]);
    }
  }
}

// ---------------------------------------------------------------------------
// Kernel 3: output projection: out[M,N] = ob[M,:] . w_proj[N,:] + b_proj[N]
// M = B*T = 4096, N = C = 1024, K = C. One wave per 32x32 block (2x2 tiles):
// 4 fragment loads : 4 WMMAs per K-step.
// Grid: 128 * 32 = 4096 waves = 512 blocks of 256 threads.
// ---------------------------------------------------------------------------
__global__ void out_proj(const bf16_t* __restrict__ ob,
                         const bf16_t* __restrict__ wpb,
                         const float* __restrict__ bias,
                         float* __restrict__ out) {
  const int lane = threadIdx.x & 31;
  const int wave = blockIdx.x * 8 + (threadIdx.x >> 5);
  const int jn = wave & 31;  // 32-col N block (C/32 = 32)
  const int im = wave >> 5;  // 32-row M block (B*T/32 = 128)
  const int m0 = im * 32;
  const int n0 = jn * 32;
  const int lrow = lane & 15;
  const int half = lane >> 4;
  const int koff = half * 8;

  const bf16_t* arow0 = ob + (size_t)(m0 + lrow) * NC;
  const bf16_t* arow1 = arow0 + (size_t)16 * NC;
  const bf16_t* brow0 = wpb + (size_t)(n0 + lrow) * NC;  // w_proj[n,k] contig k
  const bf16_t* brow1 = brow0 + (size_t)16 * NC;

  v8f acc00 = {}, acc01 = {}, acc10 = {}, acc11 = {};
  for (int kk = 0; kk < NC; kk += 32) {
    __builtin_prefetch(arow0 + kk + 128, 0, 0);
    v16bf a0 = load_frag(arow0 + kk, koff);
    v16bf a1 = load_frag(arow1 + kk, koff);
    v16bf b0 = load_frag(brow0 + kk, koff);
    v16bf b1 = load_frag(brow1 + kk, koff);
    acc00 = wmma_bf16(a0, b0, acc00);
    acc01 = wmma_bf16(a0, b1, acc01);
    acc10 = wmma_bf16(a1, b0, acc10);
    acc11 = wmma_bf16(a1, b1, acc11);
  }

  const float bv0 = bias[n0 + lrow];
  const float bv1 = bias[n0 + 16 + lrow];
#pragma unroll
  for (int r = 0; r < 8; ++r) {
    float* orow0 = out + (size_t)(m0 + r + 8 * half) * NC + n0 + lrow;
    float* orow1 = orow0 + (size_t)16 * NC;
    orow0[0] = acc00[r] + bv0;
    orow0[16] = acc01[r] + bv1;
    orow1[0] = acc10[r] + bv0;
    orow1[16] = acc11[r] + bv1;
  }
}

// ---------------------------------------------------------------------------
// Host-side launcher
// ---------------------------------------------------------------------------
extern "C" void kernel_launch(void* const* d_in, const int* in_sizes, int n_in,
                              void* d_out, int out_size, void* d_ws,
                              size_t ws_size, hipStream_t stream) {
  (void)in_sizes; (void)n_in; (void)out_size; (void)ws_size;
  const float* x = (const float*)d_in[0];       // [B,T,C]
  const float* wq = (const float*)d_in[1];      // [H,C,HS]
  const float* wk = (const float*)d_in[2];      // [H,C,HS]
  const float* wv = (const float*)d_in[3];      // [H,C,HS]
  const float* wp = (const float*)d_in[4];      // [C,C]
  const float* bp = (const float*)d_in[5];      // [C]
  float* out = (float*)d_out;                   // [B,T,C] fp32

  const size_t XN = (size_t)NB * NT * NC;       // 4,194,304
  const size_t WN = (size_t)NH * NC * ND;       // 1,048,576
  const size_t QN = (size_t)NB * NH * NT * ND;  // 4,194,304

  bf16_t* ws = (bf16_t*)d_ws;
  bf16_t* xb  = ws;             // x bf16                  [B,T,C]
  bf16_t* wqt = xb + XN;        // wq^T bf16               [H,HS,C]
  bf16_t* wkt = wqt + WN;       // wk^T bf16               [H,HS,C]
  bf16_t* wvt = wkt + WN;       // wv^T bf16               [H,HS,C]
  bf16_t* wpb = wvt + WN;       // w_proj bf16             [C,C]
  bf16_t* qb  = wpb + WN;       // Q bf16                  [B*H,T,HS]
  bf16_t* kb  = qb + QN;        // K bf16                  [B*H,T,HS]
  bf16_t* vtb = kb + QN;        // V^T bf16                [B*H,HS,T]
  bf16_t* ob  = vtb + QN;       // attn out bf16           [B,T,C]
  // total: ~48 MB of workspace

  cvt_bf16<<<(int)((XN + 255) / 256), 256, 0, stream>>>(x, xb, (int)XN);
  cvt_bf16_tr<<<(int)((WN + 255) / 256), 256, 0, stream>>>(wq, wqt, NH, NC, ND);
  cvt_bf16_tr<<<(int)((WN + 255) / 256), 256, 0, stream>>>(wk, wkt, NH, NC, ND);
  cvt_bf16_tr<<<(int)((WN + 255) / 256), 256, 0, stream>>>(wv, wvt, NH, NC, ND);
  cvt_bf16<<<(int)((WN + 255) / 256), 256, 0, stream>>>(wp, wpb, (int)WN);

  qkv_gemm<<<1024, 256, 0, stream>>>(xb, wqt, wkt, wvt, qb, kb, vtb);
  flash_attn<<<512, 256, 0, stream>>>(qb, kb, vtb, ob);
  out_proj<<<512, 256, 0, stream>>>(ob, wpb, bp, out);
}
